// BClassifier_31164282699861
// MI455X (gfx1250) — compile-verified, matrix operand order checked by reading, not run on hardware
//
#include <hip/hip_runtime.h>
#include <math.h>

typedef __attribute__((ext_vector_type(16))) __bf16 v16bf;
typedef __attribute__((ext_vector_type(8)))  float  v8f;
typedef __attribute__((ext_vector_type(4)))  unsigned int v4u;
typedef __attribute__((ext_vector_type(8)))  int v8i;
typedef __attribute__((ext_vector_type(4)))  int v4i;

// ---------- problem constants ----------
constexpr int BB = 4;
constexpr int NN = 8192;
constexpr int DM = 1024;
constexpr int DD = 512;
constexpr int RR = 4848;
constexpr int KK = 16;           // top-k
constexpr int MROWS = BB * NN;   // 32768

// GEMM tiling
constexpr int BM = 16;
constexpr int BK = 32;
constexpr int ASTRIDE = 514;     // bf16 elems per LDS A row (1024B + 4B TDM pad)
constexpr int BSTRIDE = 34;      // bf16 elems per LDS B row (64B + 4B TDM pad)

#if defined(__has_builtin)
#  if __has_builtin(__builtin_amdgcn_tensor_load_to_lds)
#    define USE_TDM 1
#  endif
#endif
#ifndef USE_TDM
#  define USE_TDM 0
#endif

__device__ __forceinline__ float gelu_exact(float x) {
    return 0.5f * x * (1.f + erff(x * 0.70710678118654752f));
}
__device__ __forceinline__ float sigmoidf_(float x) {
    return 1.f / (1.f + __expf(-x));
}
__device__ __forceinline__ unsigned lds_off(const void* p) {
    // LDS flat addresses carry the byte offset in addr[31:0]
    return (unsigned)(unsigned long long)p;
}

#if USE_TDM
// 2D Tensor-Data-Mover load: tile_d1 rows of tile_d0 elements (data_size code:
// 0=1B,1=2B,3=8B), source row stride stride_d0 elements, optional LDS padding
// of (pad_amount+1) dwords every 2^(pad_interval+1) dwords.
__device__ __forceinline__ void tdm_load_2d(unsigned lds_byte_off, const void* gsrc,
                                            unsigned ds_code,
                                            unsigned tile_d0, unsigned tile_d1,
                                            unsigned stride_d0,
                                            unsigned pad_interval, unsigned pad_amount,
                                            bool pad_en)
{
    const unsigned long long ga = (unsigned long long)(size_t)gsrc;
    v4u g0;
    g0[0] = 1u;                                        // count=1 valid descriptor
    g0[1] = lds_byte_off;                              // lds_addr
    g0[2] = (unsigned)ga;                              // global_addr[31:0]
    g0[3] = (unsigned)((ga >> 32) & 0x01ffffffu) | (2u << 30);  // addr[56:32], type=2
    v8i g1;
    g1[0] = (int)((ds_code << 16) | (pad_en ? (1u << 20) : 0u) |
                  (pad_interval << 22) | (pad_amount << 25));
    g1[1] = (int)((tile_d0 & 0xffffu) << 16);          // tensor_dim0[15:0]
    g1[2] = (int)(((tile_d0 >> 16) & 0xffffu) | ((tile_d1 & 0xffffu) << 16)); // dim0 hi | tensor_dim1 lo
    g1[3] = (int)((tile_d0 & 0xffffu) << 16);          // tensor_dim1 hi=0 | tile_dim0
    g1[4] = (int)(tile_d1 & 0xffffu);                  // tile_dim1 | tile_dim2=0
    g1[5] = (int)stride_d0;                            // tensor_dim0_stride lo32
    g1[6] = 0;
    g1[7] = 0;
    const v4i zz = {0, 0, 0, 0};
#if __clang_major__ >= 23
    const v8i z8 = {0, 0, 0, 0, 0, 0, 0, 0};
    __builtin_amdgcn_tensor_load_to_lds(g0, g1, zz, zz, z8, 0);
#else
    __builtin_amdgcn_tensor_load_to_lds(g0, g1, zz, zz, 0);
#endif
}
#endif

// Load a 16-element bf16 WMMA fragment (K-major row of 32 contiguous bf16).
// Layout per CDNA5 ISA 7.12.2 (16-bit A 16x32): element e of lane l holds
// k = 16*(e>>3) + 8*(l>=16) + (e&7).
__device__ __forceinline__ v16bf frag_load(const __bf16* __restrict__ row) {
    const int lane = threadIdx.x & 31;
    const int kh = (lane & 16) ? 8 : 0;
    v16bf f;
#pragma unroll
    for (int i = 0; i < 8; ++i) {
        const int k = ((i >> 2) << 4) + kh + ((i & 3) << 1);
        f[2 * i]     = row[k];
        f[2 * i + 1] = row[k + 1];
    }
    return f;
}

// ============ prep: fp32 -> bf16 bulk convert ============
__global__ __launch_bounds__(256)
void cvt_bf16_kernel(const float* __restrict__ x, __bf16* __restrict__ y)
{
    const size_t base = ((size_t)blockIdx.x * 256 + threadIdx.x) * 4;
    const float4 v = *(const float4*)(x + base);
    y[base + 0] = (__bf16)v.x; y[base + 1] = (__bf16)v.y;
    y[base + 2] = (__bf16)v.z; y[base + 3] = (__bf16)v.w;
}

// ============ prep: W[k][n] fp32 -> Wt[n][k] bf16 (512x512) ============
__global__ __launch_bounds__(256)
void wtrans_kernel(const float* __restrict__ W, __bf16* __restrict__ Wt)
{
    __shared__ float tile[32][33];
    const int bx = blockIdx.x, by = blockIdx.y;
    const int tx = threadIdx.x & 31, ty = threadIdx.x >> 5;   // 32 x 8
#pragma unroll
    for (int i = 0; i < 4; ++i)
        tile[ty + 8 * i][tx] = W[(size_t)(by * 32 + ty + 8 * i) * DD + bx * 32 + tx];
    __syncthreads();
#pragma unroll
    for (int i = 0; i < 4; ++i)
        Wt[(size_t)(bx * 32 + ty + 8 * i) * DD + by * 32 + tx] = (__bf16)tile[tx][ty + 8 * i];
}

// =====================================================================
// Fused  gelu(layernorm(X @ W + b)) : X bf16 [M,512], Wt bf16 [n][k].
// A panel (16x512) TDM-loaded once; 512x32 weight chunk TDM-loaded per
// k-step with hardware LDS padding (17-dword rows, conflict free).
// =====================================================================
template<bool OUT_BF16, bool OUT_F32>
__global__ __launch_bounds__(256)
void gemm_ln_gelu_kernel(const __bf16* __restrict__ x, const __bf16* __restrict__ Wt,
                         const float* __restrict__ bias,
                         const float* __restrict__ gam, const float* __restrict__ bet,
                         __bf16* __restrict__ outh, float* __restrict__ outf)
{
    __shared__ __bf16 A_s[BM][ASTRIDE];
    union UB { __bf16 B[DD][BSTRIDE]; float C[BM][DD]; };
    __shared__ UB u;
    __shared__ float bias_s[DD], gam_s[DD], bet_s[DD];
    __shared__ float red1[BM][16], red2[BM][16], mv[BM][2];

    const int tid  = threadIdx.x;
    const int lane = tid & 31;
    const int w    = tid >> 5;
    const int m0   = blockIdx.x * BM;

    for (int e = tid; e < DD; e += 256) {
        bias_s[e] = bias[e]; gam_s[e] = gam[e]; bet_s[e] = bet[e];
    }

#if USE_TDM
    if (w == 0)   // whole A panel: 16 rows x 1024B, pad rows to 1028B
        tdm_load_2d(lds_off(&A_s[0][0]), x + (size_t)m0 * DD,
                    /*8B*/3, 128, BM, 128, /*pad 1 dw / 256 dw*/7, 0, true);
#else
    for (int e = tid; e < BM * 256; e += 256) {
        const int r = e >> 8, c = e & 255;
        *(unsigned*)&A_s[r][c * 2] = *(const unsigned*)(x + (size_t)(m0 + r) * DD + c * 2);
    }
#endif

    v8f acc[4];
#pragma unroll
    for (int t = 0; t < 4; ++t)
#pragma unroll
        for (int r = 0; r < 8; ++r) acc[t][r] = 0.f;

    for (int k0 = 0; k0 < DD; k0 += BK) {
        __syncthreads();
#if USE_TDM
        if (w == 0) {   // weight chunk: 512 rows x 64B, pad rows to 68B
            tdm_load_2d(lds_off(&u.B[0][0]), Wt + k0,
                        /*2B*/1, BK, DD, DD, /*pad 1 dw / 16 dw*/3, 0, true);
            __builtin_amdgcn_s_wait_tensorcnt(0);
        }
#else
        for (int e = tid; e < DD; e += 256) {
            const __bf16* src = Wt + (size_t)e * DD + k0;
#pragma unroll
            for (int j = 0; j < 16; ++j)
                *((unsigned*)&u.B[e][0] + j) = *((const unsigned*)src + j);
        }
#endif
        __syncthreads();
        const v16bf fa = frag_load(&A_s[lane & 15][k0]);
#pragma unroll
        for (int t = 0; t < 4; ++t) {
            const v16bf fb = frag_load(&u.B[w * 64 + t * 16 + (lane & 15)][0]);
            acc[t] = __builtin_amdgcn_wmma_f32_16x16x32_bf16(
                false, fa, false, fb, (short)0, acc[t], false, false);
        }
    }
    __syncthreads();
    // accumulators + bias -> C (overlays weight staging buffer)
#pragma unroll
    for (int t = 0; t < 4; ++t) {
        const int col = w * 64 + t * 16 + (lane & 15);
        const int mb  = (lane & 16) ? 8 : 0;
#pragma unroll
        for (int r = 0; r < 8; ++r)
            u.C[r + mb][col] = acc[t][r] + bias_s[col];
    }
    __syncthreads();
    { // LayerNorm stats: 16 threads/row, 32 cols each
        const int row = tid >> 4, slot = tid & 15;
        float s1 = 0.f, s2 = 0.f;
        for (int c = slot * 32; c < slot * 32 + 32; ++c) {
            const float v = u.C[row][c]; s1 += v; s2 += v * v;
        }
        red1[row][slot] = s1; red2[row][slot] = s2;
    }
    __syncthreads();
    if ((tid & 15) == 0) {
        const int row = tid >> 4;
        float s1 = 0.f, s2 = 0.f;
        for (int i = 0; i < 16; ++i) { s1 += red1[row][i]; s2 += red2[row][i]; }
        const float mean = s1 / DD;
        const float var  = s2 / DD - mean * mean;
        mv[row][0] = mean; mv[row][1] = rsqrtf(var + 1e-5f);
    }
    __syncthreads();
    {
        const int row = tid >> 4, slot = tid & 15;
        const float mean = mv[row][0], rstd = mv[row][1];
        const size_t ob = (size_t)(m0 + row) * DD;
        for (int c = slot * 32; c < slot * 32 + 32; ++c) {
            const float xn = (u.C[row][c] - mean) * rstd * gam_s[c] + bet_s[c];
            const float y = gelu_exact(xn);
            if (OUT_BF16) outh[ob + c] = (__bf16)y;
            if (OUT_F32)  outf[ob + c] = y;
        }
    }
}

// =====================================================================
// Gated attention logits: score = (tanh(V@wa+ba)*sigmoid(V@wb+bb))@wc+bc
// V in bf16; wa/wb pre-transposed bf16; same TDM staging as GEMM.
// =====================================================================
__global__ __launch_bounds__(256)
void attn_score_kernel(const __bf16* __restrict__ Vb,
                       const __bf16* __restrict__ wta, const float* __restrict__ ba,
                       const __bf16* __restrict__ wtb, const float* __restrict__ bb,
                       const float* __restrict__ wc, const float* __restrict__ bc,
                       float* __restrict__ scores)
{
    __shared__ __bf16 A_s[BM][ASTRIDE];
    __shared__ __bf16 B_s[DD][BSTRIDE];
    __shared__ float ba_s[DD], bb_s[DD], wc_s[DD];
    __shared__ float part[8][16];

    const int tid  = threadIdx.x;
    const int lane = tid & 31;
    const int w    = tid >> 5;
    const int m0   = blockIdx.x * BM;

    for (int e = tid; e < DD; e += 256) {
        ba_s[e] = ba[e]; bb_s[e] = bb[e]; wc_s[e] = wc[e];
    }

#if USE_TDM
    if (w == 0)
        tdm_load_2d(lds_off(&A_s[0][0]), Vb + (size_t)m0 * DD, 3, 128, BM, 128, 7, 0, true);
#else
    for (int e = tid; e < BM * 256; e += 256) {
        const int r = e >> 8, c = e & 255;
        *(unsigned*)&A_s[r][c * 2] = *(const unsigned*)(Vb + (size_t)(m0 + r) * DD + c * 2);
    }
#endif

    v8f ca[4], cb[4];
#pragma unroll
    for (int t = 0; t < 4; ++t)
#pragma unroll
        for (int r = 0; r < 8; ++r) { ca[t][r] = 0.f; cb[t][r] = 0.f; }

    for (int k0 = 0; k0 < DD; k0 += BK) {
        __syncthreads();
#if USE_TDM
        if (w == 0) {
            tdm_load_2d(lds_off(&B_s[0][0]), wta + k0, 1, BK, DD, DD, 3, 0, true);
            __builtin_amdgcn_s_wait_tensorcnt(0);
        }
#else
        for (int e = tid; e < DD; e += 256) {
            const __bf16* src = wta + (size_t)e * DD + k0;
#pragma unroll
            for (int j = 0; j < 16; ++j)
                *((unsigned*)&B_s[e][0] + j) = *((const unsigned*)src + j);
        }
#endif
        __syncthreads();
        const v16bf fa = frag_load(&A_s[lane & 15][k0]);
#pragma unroll
        for (int t = 0; t < 4; ++t) {
            const v16bf fb = frag_load(&B_s[w * 64 + t * 16 + (lane & 15)][0]);
            ca[t] = __builtin_amdgcn_wmma_f32_16x16x32_bf16(
                false, fa, false, fb, (short)0, ca[t], false, false);
        }
        __syncthreads();
#if USE_TDM
        if (w == 0) {
            tdm_load_2d(lds_off(&B_s[0][0]), wtb + k0, 1, BK, DD, DD, 3, 0, true);
            __builtin_amdgcn_s_wait_tensorcnt(0);
        }
#else
        for (int e = tid; e < DD; e += 256) {
            const __bf16* src = wtb + (size_t)e * DD + k0;
#pragma unroll
            for (int j = 0; j < 16; ++j)
                *((unsigned*)&B_s[e][0] + j) = *((const unsigned*)src + j);
        }
#endif
        __syncthreads();
#pragma unroll
        for (int t = 0; t < 4; ++t) {
            const v16bf fb = frag_load(&B_s[w * 64 + t * 16 + (lane & 15)][0]);
            cb[t] = __builtin_amdgcn_wmma_f32_16x16x32_bf16(
                false, fa, false, fb, (short)0, cb[t], false, false);
        }
    }
    // epilogue: tanh * sigmoid, dot with wc, deterministic reduction
    float pr[8];
#pragma unroll
    for (int r = 0; r < 8; ++r) pr[r] = 0.f;
#pragma unroll
    for (int t = 0; t < 4; ++t) {
        const int col = w * 64 + t * 16 + (lane & 15);
#pragma unroll
        for (int r = 0; r < 8; ++r) {
            const float a = tanhf(ca[t][r] + ba_s[col]);
            const float s = sigmoidf_(cb[t][r] + bb_s[col]);
            pr[r] += a * s * wc_s[col];
        }
    }
#pragma unroll
    for (int off = 8; off; off >>= 1)
#pragma unroll
        for (int r = 0; r < 8; ++r) pr[r] += __shfl_xor(pr[r], off);
    if ((lane & 15) == 0) {
        const int mb = (lane & 16) ? 8 : 0;
#pragma unroll
        for (int r = 0; r < 8; ++r) part[w][r + mb] = pr[r];
    }
    __syncthreads();
    if (tid < BM) {
        float s = bc[0];
        for (int ww = 0; ww < 8; ++ww) s += part[ww][tid];
        scores[m0 + tid] = s;
    }
}

// ============ softmax over N per batch ============
__global__ __launch_bounds__(256)
void softmax_kernel(const float* __restrict__ scores,
                    float* __restrict__ ap_ws, float* __restrict__ ap_out)
{
    __shared__ float sv[NN];
    __shared__ float red[256];
    __shared__ float stat[2];
    const int b = blockIdx.x, tid = threadIdx.x;
    const float* sb = scores + (size_t)b * NN;
    for (int e = tid; e < NN; e += 256) sv[e] = sb[e];
    __syncthreads();
    float mx = -INFINITY;
    for (int e = tid; e < NN; e += 256) mx = fmaxf(mx, sv[e]);
    red[tid] = mx; __syncthreads();
    for (int s = 128; s; s >>= 1) { if (tid < s) red[tid] = fmaxf(red[tid], red[tid + s]); __syncthreads(); }
    if (tid == 0) stat[0] = red[0];
    __syncthreads();
    const float m = stat[0];
    float sum = 0.f;
    for (int e = tid; e < NN; e += 256) sum += __expf(sv[e] - m);
    red[tid] = sum; __syncthreads();
    for (int s = 128; s; s >>= 1) { if (tid < s) red[tid] += red[tid + s]; __syncthreads(); }
    if (tid == 0) stat[1] = 1.f / red[0];
    __syncthreads();
    const float inv = stat[1];
    for (int e = tid; e < NN; e += 256) {
        const float p = __expf(sv[e] - m) * inv;
        ap_ws[(size_t)b * NN + e] = p;
        ap_out[(size_t)b * NN + e] = p;
    }
}

// ============ B_deep = sum_n A_patch * V ============
__global__ __launch_bounds__(256)
void bdeep_kernel(const float* __restrict__ ap, const float* __restrict__ V,
                  float* __restrict__ bd_ws, float* __restrict__ bd_out)
{
    const int b = blockIdx.x;
    const int d = blockIdx.y * 256 + threadIdx.x;
    float acc = 0.f;
    const float* apb = ap + (size_t)b * NN;
    const float* vb  = V + (size_t)b * NN * DD;
    for (int n = 0; n < NN; ++n) acc += apb[n] * vb[(size_t)n * DD + d];
    bd_ws[b * DD + d] = acc;
    bd_out[b * DD + d] = acc;
}

// ============ deterministic top-16 (stable: smaller index wins ties) ============
__global__ __launch_bounds__(256)
void topk_kernel(const float* __restrict__ scores, int* __restrict__ idx)
{
    __shared__ float sv[NN];
    __shared__ float bv[256];
    __shared__ int   bi[256];
    const int b = blockIdx.x, tid = threadIdx.x;
    for (int e = tid; e < NN; e += 256) sv[e] = scores[(size_t)b * NN + e];
    __syncthreads();
    for (int kk = 0; kk < KK; ++kk) {
        float best = -INFINITY; int besti = NN;
        for (int e = tid; e < NN; e += 256) {
            const float v = sv[e];
            if (v > best || (v == best && e < besti)) { best = v; besti = e; }
        }
        bv[tid] = best; bi[tid] = besti; __syncthreads();
        for (int s = 128; s; s >>= 1) {
            if (tid < s) {
                const float v2 = bv[tid + s]; const int i2 = bi[tid + s];
                if (v2 > bv[tid] || (v2 == bv[tid] && i2 < bi[tid])) { bv[tid] = v2; bi[tid] = i2; }
            }
            __syncthreads();
        }
        if (tid == 0) { idx[b * KK + kk] = bi[0]; sv[bi[0]] = -INFINITY; }
        __syncthreads();
    }
}

// ============ gather top-k feats + sum over k (output slot 1) ============
__global__ __launch_bounds__(256)
void gather_sum_kernel(const float* __restrict__ feats, const int* __restrict__ idx,
                       float* __restrict__ topk, float* __restrict__ out1)
{
    const int b = blockIdx.x;
    const int d = blockIdx.y * 256 + threadIdx.x;
    float acc = 0.f;
    for (int kk = 0; kk < KK; ++kk) {
        const int n = idx[b * KK + kk];
        const float v = feats[((size_t)b * NN + n) * DM + d];
        topk[((size_t)b * KK + kk) * DM + d] = v;
        acc += v;
    }
    out1[(size_t)b * DM + d] = acc;
}

// ============ mixer state init: x[b][d][k] = topk[b][k][d] ============
__global__ __launch_bounds__(256)
void xinit_kernel(const float* __restrict__ topk, float* __restrict__ x)
{
    const int b = blockIdx.x;
    const int d = blockIdx.y * 256 + threadIdx.x;
#pragma unroll
    for (int kk = 0; kk < KK; ++kk)
        x[((size_t)b * DM + d) * KK + kk] = topk[((size_t)b * KK + kk) * DM + d];
}

// ============ mixer token mixing (one layer) ============
__global__ __launch_bounds__(256)
void token_mix_kernel(float* __restrict__ x,
                      const float* __restrict__ tw1, const float* __restrict__ tb1,
                      const float* __restrict__ tw2, const float* __restrict__ tb2)
{
    __shared__ float t1[KK][DD];   // 32 KB
    const int b = blockIdx.x, tid = threadIdx.x;
    const float* xb = x + (size_t)b * DM * KK;
    for (int o = 0; o < 32; ++o) {
        const int id = tid + (o << 8);
        const int r = id >> 9, jj = id & 511;
        float acc = tb1[jj];
        for (int d = 0; d < DM; ++d)
            acc += xb[d * KK + r] * tw1[(size_t)d * DD + jj];
        t1[r][jj] = gelu_exact(acc);
    }
    __syncthreads();
    for (int o = 0; o < 64; ++o) {
        const int id = tid + (o << 8);
        const int r = id >> 10, dm = id & 1023;
        float acc = tb2[dm];
        for (int jj = 0; jj < DD; ++jj)
            acc += t1[r][jj] * tw2[(size_t)jj * DM + dm];
        x[((size_t)b * DM + dm) * KK + r] += acc;
    }
}

// ============ mixer channel mixing (one layer) ============
__global__ __launch_bounds__(256)
void channel_mix_kernel(float* __restrict__ x,
                        const float* __restrict__ cw1, const float* __restrict__ cb1,
                        const float* __restrict__ cw2, const float* __restrict__ cb2)
{
    const int b = blockIdx.x;
    const int dm = blockIdx.y * 256 + threadIdx.x;
    float* xr = x + ((size_t)b * DM + dm) * KK;
    float xv[KK], acc[KK];
#pragma unroll
    for (int k = 0; k < KK; ++k) { xv[k] = xr[k]; acc[k] = cb2[k]; }
    for (int jj = 0; jj < DD; ++jj) {
        float u = cb1[jj];
#pragma unroll
        for (int k = 0; k < KK; ++k) u += xv[k] * cw1[k * DD + jj];
        u = gelu_exact(u);
#pragma unroll
        for (int k = 0; k < KK; ++k) acc[k] += u * cw2[jj * KK + k];
    }
#pragma unroll
    for (int k = 0; k < KK; ++k) xr[k] = xv[k] + acc[k];
}

// ============ A_feat gating; also out0 = A_feat * sum(topk) ============
__global__ __launch_bounds__(256)
void afeat_kernel(const float* __restrict__ x,
                  const float* __restrict__ wa, const float* __restrict__ ba,
                  const float* __restrict__ wb, const float* __restrict__ bb,
                  const float* __restrict__ wc, const float* __restrict__ bc,
                  float* __restrict__ out)
{
    const int b = blockIdx.x;
    const int dm = blockIdx.y * 256 + threadIdx.x;
    const float* xr = x + ((size_t)b * DM + dm) * KK;
    float xv[KK];
#pragma unroll
    for (int k = 0; k < KK; ++k) xv[k] = xr[k];
    float acc = bc[0];
    for (int jj = 0; jj < DD; ++jj) {
        float da = ba[jj], db = bb[jj];
#pragma unroll
        for (int k = 0; k < KK; ++k) {
            da += xv[k] * wa[k * DD + jj];
            db += xv[k] * wb[k * DD + jj];
        }
        acc += tanhf(da) * sigmoidf_(db) * wc[jj];
    }
    const float af = sigmoidf_(acc);
    out[20480 + (size_t)b * DM + dm] = af;                               // slot6: A_feat
    out[(size_t)b * DM + dm] = af * out[4096 + (size_t)b * DM + dm];     // slot0
}

// ============ RNA GEMV: y[b][col] = x[b]·W[:,col] + bias, M=4 ============
__global__ __launch_bounds__(256)
void rna_gemv_kernel(const float* __restrict__ x, const float* __restrict__ W,
                     const float* __restrict__ bias, float* __restrict__ y,
                     int Kdim, int Ncols, float* __restrict__ y2)
{
    __shared__ float xs[BB][1024];
    const int tid = threadIdx.x;
    const int col = blockIdx.x * 256 + tid;
    float acc[BB] = {0.f, 0.f, 0.f, 0.f};
    for (int k0 = 0; k0 < Kdim; k0 += 1024) {
        const int cnt = min(1024, Kdim - k0);
        __syncthreads();
        for (int e = tid; e < BB * cnt; e += 256) {
            const int bb2 = e & 3, kk = e >> 2;
            xs[bb2][kk] = x[(size_t)bb2 * Kdim + k0 + kk];
        }
        __syncthreads();
        if (col < Ncols) {
            for (int k = 0; k < cnt; ++k) {
                const float wv = W[(size_t)(k0 + k) * Ncols + col];
#pragma unroll
                for (int bb2 = 0; bb2 < BB; ++bb2) acc[bb2] += xs[bb2][k] * wv;
            }
        }
    }
    if (col < Ncols) {
#pragma unroll
        for (int bb2 = 0; bb2 < BB; ++bb2) {
            const float v = acc[bb2] + bias[col];
            y[(size_t)bb2 * Ncols + col] = v;
            if (y2) y2[(size_t)bb2 * Ncols + col] = v;
        }
    }
}

// ============ LayerNorm + ReLU over length-n rows ============
__global__ __launch_bounds__(256)
void ln_relu_kernel(const float* __restrict__ x, const float* __restrict__ g,
                    const float* __restrict__ be, float* __restrict__ y, int n)
{
    __shared__ float r1[256], r2[256], mv[2];
    const int b = blockIdx.x, tid = threadIdx.x;
    const float* xb = x + (size_t)b * n;
    float s1 = 0.f, s2 = 0.f;
    for (int e = tid; e < n; e += 256) { const float v = xb[e]; s1 += v; s2 += v * v; }
    r1[tid] = s1; r2[tid] = s2; __syncthreads();
    for (int s = 128; s; s >>= 1) {
        if (tid < s) { r1[tid] += r1[tid + s]; r2[tid] += r2[tid + s]; }
        __syncthreads();
    }
    if (tid == 0) {
        const float m = r1[0] / n;
        mv[0] = m; mv[1] = rsqrtf(r2[0] / n - m * m + 1e-5f);
    }
    __syncthreads();
    const float m = mv[0], rs = mv[1];
    for (int e = tid; e < n; e += 256) {
        const float v = (xb[e] - m) * rs * g[e] + be[e];
        y[(size_t)b * n + e] = fmaxf(v, 0.f);
    }
}

// ============ final projections: [4,512] @ [512,1024] + b ============
__global__ __launch_bounds__(256)
void proj_kernel(const float* __restrict__ vec, const float* __restrict__ W,
                 const float* __restrict__ bias, float* __restrict__ y)
{
    const int flat = blockIdx.x * 256 + threadIdx.x;     // 0..4095
    const int b = flat >> 10, n = flat & 1023;
    float acc = bias[n];
    for (int k = 0; k < DD; ++k) acc += vec[b * DD + k] * W[(size_t)k * DM + n];
    y[flat] = acc;
}

// =====================================================================
extern "C" void kernel_launch(void* const* d_in, const int* in_sizes, int n_in,
                              void* d_out, int out_size, void* d_ws, size_t ws_size,
                              hipStream_t stream)
{
    const float* feats      = (const float*)d_in[0];
    const float* feats_deep = (const float*)d_in[1];
    const float* rna_seq    = (const float*)d_in[2];
    const float* pa_w1 = (const float*)d_in[3];  const float* pa_b1 = (const float*)d_in[4];
    const float* pa_g1 = (const float*)d_in[5];  const float* pa_be1 = (const float*)d_in[6];
    const float* pa_w2 = (const float*)d_in[7];  const float* pa_b2 = (const float*)d_in[8];
    const float* pa_g2 = (const float*)d_in[9];  const float* pa_be2 = (const float*)d_in[10];
    const float* ad_wa = (const float*)d_in[11]; const float* ad_ba = (const float*)d_in[12];
    const float* ad_wb = (const float*)d_in[13]; const float* ad_bb = (const float*)d_in[14];
    const float* ad_wc = (const float*)d_in[15]; const float* ad_bc = (const float*)d_in[16];
    const float* ag_wa = (const float*)d_in[17]; const float* ag_ba = (const float*)d_in[18];
    const float* ag_wb = (const float*)d_in[19]; const float* ag_bb = (const float*)d_in[20];
    const float* ag_wc = (const float*)d_in[21]; const float* ag_bc = (const float*)d_in[22];
    const float* mx_tw1 = (const float*)d_in[23]; const float* mx_tb1 = (const float*)d_in[24];
    const float* mx_tw2 = (const float*)d_in[25]; const float* mx_tb2 = (const float*)d_in[26];
    const float* mx_cw1 = (const float*)d_in[27]; const float* mx_cb1 = (const float*)d_in[28];
    const float* mx_cw2 = (const float*)d_in[29]; const float* mx_cb2 = (const float*)d_in[30];
    const float* rn_w1 = (const float*)d_in[31]; const float* rn_b1 = (const float*)d_in[32];
    const float* rn_g1 = (const float*)d_in[33]; const float* rn_be1 = (const float*)d_in[34];
    const float* rn_w2 = (const float*)d_in[35]; const float* rn_b2 = (const float*)d_in[36];
    const float* rn_g2 = (const float*)d_in[37]; const float* rn_be2 = (const float*)d_in[38];
    const float* rn_w3 = (const float*)d_in[39]; const float* rn_b3 = (const float*)d_in[40];
    const float* cp_w = (const float*)d_in[41]; const float* cp_b = (const float*)d_in[42];
    const float* cr_w = (const float*)d_in[43]; const float* cr_b = (const float*)d_in[44];

    float* out = (float*)d_out;

    // workspace carve-up (256B aligned)
    char* ws = (char*)d_ws;
    size_t off = 0;
    auto carve = [&](size_t bytes) -> char* {
        char* p = ws + off;
        off = (off + bytes + 255) & ~(size_t)255;
        return p;
    };
    __bf16* fd_bf = (__bf16*)carve((size_t)MROWS * DD * sizeof(__bf16));
    __bf16* h_bf  = (__bf16*)carve((size_t)MROWS * DD * sizeof(__bf16));
    __bf16* v_bf  = (__bf16*)carve((size_t)MROWS * DD * sizeof(__bf16));
    float*  V_f   = (float*)carve((size_t)MROWS * DD * sizeof(float));
    __bf16* wt1   = (__bf16*)carve((size_t)DD * DD * sizeof(__bf16));
    __bf16* wt2   = (__bf16*)carve((size_t)DD * DD * sizeof(__bf16));
    __bf16* wta   = (__bf16*)carve((size_t)DD * DD * sizeof(__bf16));
    __bf16* wtb   = (__bf16*)carve((size_t)DD * DD * sizeof(__bf16));
    float*  sc_ws = (float*)carve((size_t)BB * NN * sizeof(float));
    float*  ap_ws = (float*)carve((size_t)BB * NN * sizeof(float));
    int*    idx_ws= (int*)carve((size_t)BB * KK * sizeof(int));
    float*  tk_ws = (float*)carve((size_t)BB * KK * DM * sizeof(float));
    float*  x_ws  = (float*)carve((size_t)BB * DM * KK * sizeof(float));
    float*  bd_ws = (float*)carve((size_t)BB * DD * sizeof(float));
    float*  r0_ws = (float*)carve((size_t)BB * RR * sizeof(float));
    float*  r1_ws = (float*)carve((size_t)BB * RR * sizeof(float));
    float*  re_ws = (float*)carve((size_t)BB * DD * sizeof(float));

    const int gemm_blocks = MROWS / BM;   // 2048
    const dim3 tgrid(16, 16);

    // prep: bf16 activations + transposed bf16 weights
    cvt_bf16_kernel<<<(MROWS * DD) / 1024, 256, 0, stream>>>(feats_deep, fd_bf);
    wtrans_kernel<<<tgrid, 256, 0, stream>>>(pa_w1, wt1);
    wtrans_kernel<<<tgrid, 256, 0, stream>>>(pa_w2, wt2);
    wtrans_kernel<<<tgrid, 256, 0, stream>>>(ad_wa, wta);
    wtrans_kernel<<<tgrid, 256, 0, stream>>>(ad_wb, wtb);

    // pre_attn layer 1: h = gelu(ln(feats_deep @ pa_w1 + b))  -> bf16
    gemm_ln_gelu_kernel<true, false><<<gemm_blocks, 256, 0, stream>>>(
        fd_bf, wt1, pa_b1, pa_g1, pa_be1, h_bf, nullptr);
    // pre_attn layer 2: V = gelu(ln(h @ pa_w2 + b))  -> bf16 + f32
    gemm_ln_gelu_kernel<true, true><<<gemm_blocks, 256, 0, stream>>>(
        h_bf, wt2, pa_b2, pa_g2, pa_be2, v_bf, V_f);
    // gated attention logits
    attn_score_kernel<<<gemm_blocks, 256, 0, stream>>>(
        v_bf, wta, ad_ba, wtb, ad_bb, ad_wc, ad_bc, sc_ws);
    // softmax -> A_patch (ws + output slot 7 @ 24576)
    softmax_kernel<<<BB, 256, 0, stream>>>(sc_ws, ap_ws, out + 24576);
    // B_deep (ws + output slot 3 @ 12288)
    bdeep_kernel<<<dim3(BB, DD / 256), 256, 0, stream>>>(ap_ws, V_f, bd_ws, out + 12288);
    // top-16 indices
    topk_kernel<<<BB, 256, 0, stream>>>(sc_ws, idx_ws);
    // gather + sum(topk) (output slot 1 @ 4096)
    gather_sum_kernel<<<dim3(BB, DM / 256), 256, 0, stream>>>(feats, idx_ws, tk_ws, out + 4096);
    // mixer init
    xinit_kernel<<<dim3(BB, DM / 256), 256, 0, stream>>>(tk_ws, x_ws);
    // 4 mixer layers
    for (int i = 0; i < 4; ++i) {
        token_mix_kernel<<<BB, 256, 0, stream>>>(
            x_ws, mx_tw1 + (size_t)i * DM * DD, mx_tb1 + (size_t)i * DD,
            mx_tw2 + (size_t)i * DD * DM, mx_tb2 + (size_t)i * DM);
        channel_mix_kernel<<<dim3(BB, DM / 256), 256, 0, stream>>>(
            x_ws, mx_cw1 + (size_t)i * KK * DD, mx_cb1 + (size_t)i * DD,
            mx_cw2 + (size_t)i * DD * KK, mx_cb2 + (size_t)i * KK);
    }
    // A_feat (slot 6 @ 20480) and out0 = A_feat * sum(topk) (slot 0 @ 0)
    afeat_kernel<<<dim3(BB, DM / 256), 256, 0, stream>>>(
        x_ws, ag_wa, ag_ba, ag_wb, ag_bb, ag_wc, ag_bc, out);
    // RNA embedder
    const int rn_blocks = (RR + 255) / 256;  // 19
    rna_gemv_kernel<<<rn_blocks, 256, 0, stream>>>(rna_seq, rn_w1, rn_b1, r0_ws, RR, RR, nullptr);
    ln_relu_kernel<<<BB, 256, 0, stream>>>(r0_ws, rn_g1, rn_be1, r1_ws, RR);
    rna_gemv_kernel<<<rn_blocks, 256, 0, stream>>>(r1_ws, rn_w2, rn_b2, r0_ws, RR, RR, nullptr);
    ln_relu_kernel<<<BB, 256, 0, stream>>>(r0_ws, rn_g2, rn_be2, r1_ws, RR);
    // rna_emb (ws + output slot 5 @ 18432)
    rna_gemv_kernel<<<(DD + 255) / 256, 256, 0, stream>>>(r1_ws, rn_w3, rn_b3, re_ws, RR, DD, out + 18432);
    // final projections: slot 2 @ 8192 (B_deep@cp), slot 4 @ 14336 (rna_emb@cr)
    proj_kernel<<<BB * DM / 256, 256, 0, stream>>>(bd_ws, cp_w, cp_b, out + 8192);
    proj_kernel<<<BB * DM / 256, 256, 0, stream>>>(re_ws, cr_w, cr_b, out + 14336);
}
